// ReconstructionLoss_56796647522360
// MI455X (gfx1250) — matrix operand order repaired and checked by reading, not compiled, not run on hardware
//
#include <hip/hip_runtime.h>

typedef __attribute__((ext_vector_type(16))) _Float16 v16h;
typedef __attribute__((ext_vector_type(8)))  float    v8f;

union Frag16 { v16h v; unsigned u[8]; };

#define CONV_WAVES 4

// ---------------------------------------------------------------------------
// Implicit-GEMM 3x3 conv (pad=1) + bias + ReLU, f16 in / f16 out, f32 accum.
// GEMM: M = 4*Hout*Wout (output pixels), N = Cout, K = Cin*9 (padded to Kp).
// Block = 4 waves = 64(M) x 64(N) output tile.
//   - B tile (64 co x 32 k) staged in LDS once per block per K-chunk via
//     async global->LDS copies (ASYNCcnt), shared by all 4 waves.
//   - Each wave gathers its own 16x32 im2col A tile, then reuses one A
//     fragment against 4 B fragments -> 4x V_WMMA_F32_16X16X32_F16 per chunk.
// All layers here have M % 64 == 0 and Cout % 64 == 0: no partial tiles,
// so block barriers are uniformly reached.
// ---------------------------------------------------------------------------
__global__ __launch_bounds__(CONV_WAVES * 32)
void conv_wmma_kernel(const _Float16* __restrict__ in,
                      const _Float16* __restrict__ wp,   // packed [Cout][Kp]
                      const float*    __restrict__ bias,
                      _Float16*       __restrict__ out,
                      int Cin, int Hin, int Win,
                      int Cout, int Hout, int Wout,
                      int stride, int K, int Kp, int relu)
{
    __shared__ _Float16 ldsA[CONV_WAVES * 512];  // per-wave 16x32 A tile
    __shared__ _Float16 ldsB[2048];              // shared 64x32 B tile
    const int tid  = threadIdx.x;
    const int wave = tid >> 5;
    const int lane = tid & 31;
    _Float16* A_lds = &ldsA[wave * 512];
    unsigned* A32 = (unsigned*)A_lds;
    unsigned* B32 = (unsigned*)ldsB;

    const int M      = 4 * Hout * Wout;
    const int coBase = blockIdx.x * 64;
    const int mBase  = (blockIdx.y * CONV_WAVES + wave) * 16;

    const unsigned ldsB_base = (unsigned)(unsigned long long)(&ldsB[0]);

    v8f cf[4] = {};

    const int nChunks = Kp >> 5;
    for (int kc = 0; kc < nChunks; ++kc) {
        const int k0 = kc << 5;

        // ---- async-copy B tile: 64 co x 32 k = 1024 dwords, 8 per thread ----
        #pragma unroll
        for (int t = 0; t < 8; ++t) {
            const int idx = t * 128 + tid;       // dword index 0..1023
            const int row = idx >> 4;            // co within tile (0..63)
            const int cdw = idx & 15;            // dword (2 halves) within row
            const unsigned gbyte =
                (unsigned)(((size_t)(coBase + row) * Kp + k0 + cdw * 2) * 2);
            const unsigned lbyte = ldsB_base + (unsigned)(idx * 4);
            asm volatile("global_load_async_to_lds_b32 %0, %1, %2"
                         :: "v"(lbyte), "v"(gbyte), "s"(wp) : "memory");
        }
        if (kc + 1 < nChunks)
            __builtin_prefetch(wp + (size_t)coBase * Kp + k0 + 32, 0, 1);

        // ---- gather A tile: 16 rows (output pixels) x 32 k (im2col) ----
        {
            const int kg = k0 + lane;
            const int ci = kg / 9;
            const int r  = kg - ci * 9;
            const int ky = r / 3;
            const int kx = r - ky * 3;
            const bool kvalid = (kg < K);
            #pragma unroll 4
            for (int t = 0; t < 16; ++t) {
                const int m = mBase + t;
                _Float16 v = (_Float16)0.0f;
                if (kvalid && m < M) {
                    const int x   = m % Wout;
                    const int tmp = m / Wout;
                    const int y   = tmp % Hout;
                    const int b   = tmp / Hout;
                    const int iy  = y * stride + ky - 1;
                    const int ix  = x * stride + kx - 1;
                    if ((unsigned)iy < (unsigned)Hin && (unsigned)ix < (unsigned)Win)
                        v = in[(((size_t)b * Cin + ci) * Hin + iy) * Win + ix];
                }
                A_lds[t * 32 + lane] = v;
            }
        }

        asm volatile("s_wait_dscnt 0\n\t"
                     "s_wait_asynccnt 0" ::: "memory");
        __syncthreads();

        // ---- build fragments per ISA 16-bit layouts (CDNA5 ISA) ----
        const int mrow  = lane & 15;             // A row M / B col N
        const int khalf = lane >> 4;
        Frag16 af;
        #pragma unroll
        for (int j = 0; j < 8; ++j) {
            // A 16x32: lanes 0-15: V0..3 K=0..7, V4..7 K=16..23; lanes 16-31: +8
            const int kkA = ((j < 4) ? (2 * j) : (16 + 2 * (j - 4))) + khalf * 8;
            af.u[j] = A32[mrow * 16 + (kkA >> 1)];
        }
        #pragma unroll
        for (int n = 0; n < 4; ++n) {
            Frag16 bf;
            #pragma unroll
            for (int j = 0; j < 8; ++j) {
                // B 32x16: lanes 0-15: V0..3 K=0..7, V4..7 K=16..23; lanes 16-31: +8
                const int kkB = ((j < 4) ? (2 * j) : (16 + 2 * (j - 4))) + khalf * 8;
                bf.u[j] = B32[(n * 16 + mrow) * 16 + (kkB >> 1)];
            }
            cf[n] = __builtin_amdgcn_wmma_f32_16x16x32_f16(
                false, af.v, false, bf.v, (short)0, cf[n], false, false);
        }
        __syncthreads();                         // protect LDS for next chunk
    }

    // ---- epilogue: C layout lanes 0-15 -> M=j, lanes 16-31 -> M=j+8 ----
    const int ncol  = lane & 15;
    const int khalf = lane >> 4;
    float bv[4];
    #pragma unroll
    for (int n = 0; n < 4; ++n)
        bv[n] = bias ? bias[coBase + n * 16 + ncol] : 0.0f;
    const size_t chStride = (size_t)Hout * Wout;
    #pragma unroll
    for (int j = 0; j < 8; ++j) {
        const int m = mBase + j + khalf * 8;
        if (m < M) {
            const int x   = m % Wout;
            const int tmp = m / Wout;
            const int y   = tmp % Hout;
            const int b   = tmp / Hout;
            const size_t o0 = (((size_t)b * Cout) * Hout + y) * Wout + x;
            #pragma unroll
            for (int n = 0; n < 4; ++n) {
                const int co = coBase + n * 16 + ncol;
                float v = cf[n][j] + bv[n];
                if (relu && v < 0.0f) v = 0.0f;
                out[o0 + (size_t)co * chStride] = (_Float16)v;
            }
        }
    }
}

// ---------------------------------------------------------------------------
// Support kernels
// ---------------------------------------------------------------------------
__global__ void zero_acc_kernel(float* acc) {
    if (blockIdx.x == 0 && threadIdx.x < 16) acc[threadIdx.x] = 0.0f;
}

// f32 [4,3,128,128] -> f16 [4,6,128,128] (channel-duplicated concat)
__global__ void cvt_cat6_kernel(const float* __restrict__ src,
                                _Float16* __restrict__ dst, int total) {
    int i = blockIdx.x * blockDim.x + threadIdx.x;
    if (i >= total) return;
    int pix = i & 16383;
    int c6  = (i >> 14) % 6;
    int b   = i / (6 * 16384);
    dst[i] = (_Float16)src[((size_t)b * 3 + (c6 % 3)) * 16384 + pix];
}

// f32 [4,3,128,128] -> normalized f16 for VGG
__global__ void cvt_vgg_kernel(const float* __restrict__ src,
                               _Float16* __restrict__ dst, int total) {
    int i = blockIdx.x * blockDim.x + threadIdx.x;
    if (i >= total) return;
    int c = (i >> 14) % 3;
    float mean = (c == 0) ? 0.485f : (c == 1) ? 0.456f : 0.406f;
    float stdv = (c == 0) ? 0.229f : (c == 1) ? 0.224f : 0.225f;
    dst[i] = (_Float16)((src[i] - mean) / stdv);
}

// f32 weights [Cout][Cin][3][3] -> f16 packed [Cout][Kp], zero pad k in [K,Kp)
__global__ void pack_w_kernel(const float* __restrict__ w,
                              _Float16* __restrict__ wp,
                              int K, int Kp, int total) {
    int i = blockIdx.x * blockDim.x + threadIdx.x;
    if (i >= total) return;
    int co = i / Kp;
    int k  = i - co * Kp;
    wp[i] = (k < K) ? (_Float16)w[(size_t)co * K + k] : (_Float16)0.0f;
}

__global__ void maxpool_kernel(const _Float16* __restrict__ in,
                               _Float16* __restrict__ out,
                               int BC, int Hi, int Wi) {
    int Ho = Hi >> 1, Wo = Wi >> 1;
    int total = BC * Ho * Wo;
    int i = blockIdx.x * blockDim.x + threadIdx.x;
    if (i >= total) return;
    int x = i % Wo;
    int t = i / Wo;
    int y = t % Ho;
    int c = t / Ho;
    const _Float16* p = in + ((size_t)c * Hi + 2 * y) * Wi + 2 * x;
    float a  = (float)p[0],  b = (float)p[1];
    float c2 = (float)p[Wi], d = (float)p[Wi + 1];
    out[i] = (_Float16)fmaxf(fmaxf(a, b), fmaxf(c2, d));
}

__device__ inline void hist_add(float* h, float x) {
    const float bw   = 1.0f / 255.0f;
    const float invd = 1.0f / (bw + 1e-6f);
    float tpos = x * 255.0f;
    float fl   = floorf(tpos);
    int   i0   = (int)fl;
    float frac = tpos - fl;
    float w0 = 1.0f - frac * bw * invd;
    float w1 = 1.0f - (1.0f - frac) * bw * invd;
    if ((unsigned)i0 < 256u && w0 > 0.0f)        atomicAdd(&h[i0], w0);
    if ((unsigned)(i0 + 1) < 256u && w1 > 0.0f)  atomicAdd(&h[i0 + 1], w1);
}

// One block per (b, c6): soft histogram of pred & tgt, cdf L1 -> acc[2]
__global__ __launch_bounds__(256)
void hist_kernel(const float* __restrict__ pred,
                 const float* __restrict__ tgt, float* acc) {
    __shared__ float h1[256], h2[256];
    const int b   = blockIdx.x / 6;
    const int c   = (blockIdx.x % 6) % 3;
    const int tid = threadIdx.x;
    h1[tid] = 0.0f; h2[tid] = 0.0f;
    __syncthreads();
    const float* p = pred + ((size_t)b * 3 + c) * 16384;
    const float* q = tgt  + ((size_t)b * 3 + c) * 16384;
    for (int i = tid; i < 16384; i += 256) {
        hist_add(h1, p[i]);
        hist_add(h2, q[i]);
    }
    __syncthreads();
    if (tid == 0) {
        float s1 = 0.0f, s2 = 0.0f, d = 0.0f;
        for (int i = 0; i < 256; ++i) { s1 += h1[i]; s2 += h2[i]; d += fabsf(s1 - s2); }
        atomicAdd(&acc[2], d / 16384.0f);
    }
}

// One block per spatial position of encoder feats [4,512,8,8]:
// accumulates MSE partial -> acc[0], cosine term -> acc[1]
__global__ __launch_bounds__(256)
void msecos_kernel(const _Float16* __restrict__ fp,
                   const _Float16* __restrict__ ft, float* acc) {
    __shared__ float s0[256], s1[256], s2[256], s3[256];
    const int p = blockIdx.x;                 // 0..255
    const int x = p & 7, y = (p >> 3) & 7, b = p >> 6;
    const size_t base = (size_t)b * 512 * 64 + (size_t)y * 8 + x;
    float num = 0.0f, na = 0.0f, nb = 0.0f, ms = 0.0f;
    for (int ch = threadIdx.x; ch < 512; ch += 256) {
        float a = (float)fp[base + (size_t)ch * 64];
        float t = (float)ft[base + (size_t)ch * 64];
        num += a * t; na += a * a; nb += t * t;
        float d = a - t; ms += d * d;
    }
    s0[threadIdx.x] = num; s1[threadIdx.x] = na;
    s2[threadIdx.x] = nb;  s3[threadIdx.x] = ms;
    __syncthreads();
    for (int st = 128; st > 0; st >>= 1) {
        if (threadIdx.x < st) {
            s0[threadIdx.x] += s0[threadIdx.x + st];
            s1[threadIdx.x] += s1[threadIdx.x + st];
            s2[threadIdx.x] += s2[threadIdx.x + st];
            s3[threadIdx.x] += s3[threadIdx.x + st];
        }
        __syncthreads();
    }
    if (threadIdx.x == 0) {
        float n1 = fmaxf(sqrtf(s1[0]), 1e-8f);
        float n2 = fmaxf(sqrtf(s2[0]), 1e-8f);
        atomicAdd(&acc[0], s3[0]);
        atomicAdd(&acc[1], s0[0] / (n1 * n2));
    }
}

__global__ __launch_bounds__(256)
void sqdiff_kernel(const _Float16* __restrict__ a,
                   const _Float16* __restrict__ b, float* acc, int total) {
    __shared__ float s[256];
    int i = blockIdx.x * 256 + threadIdx.x;
    float d = 0.0f;
    if (i < total) { float x = (float)a[i] - (float)b[i]; d = x * x; }
    s[threadIdx.x] = d;
    __syncthreads();
    for (int st = 128; st > 0; st >>= 1) {
        if (threadIdx.x < st) s[threadIdx.x] += s[threadIdx.x + st];
        __syncthreads();
    }
    if (threadIdx.x == 0) atomicAdd(&acc[3], s[0]);
}

__global__ void finalize_kernel(const float* acc, float* out) {
    out[0] = acc[0] / 131072.0f                    // ALPHA * mse
           + (1.0f - acc[1] / 256.0f)              // GAMMA * cosine
           + acc[2] / 6144.0f                      // BETA  * hist
           + 0.1f * (acc[3] / 524288.0f);          // DELTA * vgg
}

// ---------------------------------------------------------------------------
// Host driver
// ---------------------------------------------------------------------------
extern "C" void kernel_launch(void* const* d_in, const int* in_sizes, int n_in,
                              void* d_out, int out_size, void* d_ws, size_t ws_size,
                              hipStream_t stream) {
    (void)in_sizes; (void)n_in; (void)out_size;

    const float* pred = (const float*)d_in[0];
    const float* tgt  = (const float*)d_in[1];
    const float* enc_w[4]; const float* enc_b[4];
    const float* vgg_w[9]; const float* vgg_b[9];
    for (int j = 0; j < 4; ++j) { enc_w[j] = (const float*)d_in[2 + j];  enc_b[j] = (const float*)d_in[6 + j]; }
    for (int j = 0; j < 9; ++j) { vgg_w[j] = (const float*)d_in[10 + j]; vgg_b[j] = (const float*)d_in[19 + j]; }

    char* ws = (char*)d_ws;
    size_t off = 0;
    auto alloc = [&](size_t bytes) -> char* {
        char* p = ws + off;
        off += (bytes + 255) & ~(size_t)255;
        return p;
    };

    float*    acc   = (float*)alloc(64);
    _Float16* PRED6 = (_Float16*)alloc((size_t)4 * 6 * 16384 * 2);
    _Float16* TGT6  = (_Float16*)alloc((size_t)4 * 6 * 16384 * 2);
    _Float16* VGGP  = (_Float16*)alloc((size_t)4 * 3 * 16384 * 2);
    _Float16* VGGT  = (_Float16*)alloc((size_t)4 * 3 * 16384 * 2);
    _Float16* XA    = (_Float16*)alloc((size_t)4 * 64 * 128 * 128 * 2);
    _Float16* XB    = (_Float16*)alloc((size_t)4 * 64 * 128 * 128 * 2);
    _Float16* FP    = (_Float16*)alloc((size_t)131072 * 2);
    _Float16* FT    = (_Float16*)alloc((size_t)131072 * 2);
    _Float16* VFP   = (_Float16*)alloc((size_t)524288 * 2);

    const int ENC_CH[5] = {6, 64, 128, 256, 512};
    const int VGG_CI[9] = {3, 64, 64, 128, 128, 256, 256, 256, 256};
    const int VGG_CO[9] = {64, 64, 128, 128, 256, 256, 256, 256, 512};

    _Float16* wpk_enc[4];
    for (int j = 0; j < 4; ++j) {
        int Kp = ((ENC_CH[j] * 9) + 31) & ~31;
        wpk_enc[j] = (_Float16*)alloc((size_t)ENC_CH[j + 1] * Kp * 2);
    }
    _Float16* wpk_vgg[9];
    for (int j = 0; j < 9; ++j) {
        int Kp = ((VGG_CI[j] * 9) + 31) & ~31;
        wpk_vgg[j] = (_Float16*)alloc((size_t)VGG_CO[j] * Kp * 2);
    }
    if (off > ws_size) return;   // workspace too small: bail out safely

    zero_acc_kernel<<<1, 32, 0, stream>>>(acc);

    const int tot6 = 4 * 6 * 16384;
    const int tot3 = 4 * 3 * 16384;
    cvt_cat6_kernel<<<(tot6 + 255) / 256, 256, 0, stream>>>(pred, PRED6, tot6);
    cvt_cat6_kernel<<<(tot6 + 255) / 256, 256, 0, stream>>>(tgt,  TGT6,  tot6);
    cvt_vgg_kernel<<<(tot3 + 255) / 256, 256, 0, stream>>>(pred, VGGP, tot3);
    cvt_vgg_kernel<<<(tot3 + 255) / 256, 256, 0, stream>>>(tgt,  VGGT, tot3);

    for (int j = 0; j < 4; ++j) {
        int K = ENC_CH[j] * 9, Kp = (K + 31) & ~31;
        int total = ENC_CH[j + 1] * Kp;
        pack_w_kernel<<<(total + 255) / 256, 256, 0, stream>>>(enc_w[j], wpk_enc[j], K, Kp, total);
    }
    for (int j = 0; j < 9; ++j) {
        int K = VGG_CI[j] * 9, Kp = (K + 31) & ~31;
        int total = VGG_CO[j] * Kp;
        pack_w_kernel<<<(total + 255) / 256, 256, 0, stream>>>(vgg_w[j], wpk_vgg[j], K, Kp, total);
    }

    auto conv = [&](const _Float16* cin_buf, _Float16* cout_buf, const _Float16* wpk,
                    const float* bias, int Cin, int Hin, int Win, int Cout, int stride) {
        int Hout = Hin / stride, Wout = Win / stride;
        int K = Cin * 9, Kp = (K + 31) & ~31;
        int M = 4 * Hout * Wout;                 // always a multiple of 64 here
        dim3 grid(Cout / 64, (M + 63) / 64);
        conv_wmma_kernel<<<grid, CONV_WAVES * 32, 0, stream>>>(
            cin_buf, wpk, bias, cout_buf,
            Cin, Hin, Win, Cout, Hout, Wout, stride, K, Kp, 1);
    };
    auto pool = [&](const _Float16* pin, _Float16* pout, int C, int Hi) {
        int total = 4 * C * (Hi / 2) * (Hi / 2);
        maxpool_kernel<<<(total + 255) / 256, 256, 0, stream>>>(pin, pout, 4 * C, Hi, Hi);
    };

    // ---- encoder (pred) ----
    conv(PRED6, XA, wpk_enc[0], enc_b[0],   6, 128, 128,  64, 2);
    conv(XA,    XB, wpk_enc[1], enc_b[1],  64,  64,  64, 128, 2);
    conv(XB,    XA, wpk_enc[2], enc_b[2], 128,  32,  32, 256, 2);
    conv(XA,    FP, wpk_enc[3], enc_b[3], 256,  16,  16, 512, 2);
    // ---- encoder (target) ----
    conv(TGT6,  XA, wpk_enc[0], enc_b[0],   6, 128, 128,  64, 2);
    conv(XA,    XB, wpk_enc[1], enc_b[1],  64,  64,  64, 128, 2);
    conv(XB,    XA, wpk_enc[2], enc_b[2], 128,  32,  32, 256, 2);
    conv(XA,    FT, wpk_enc[3], enc_b[3], 256,  16,  16, 512, 2);

    msecos_kernel<<<256, 256, 0, stream>>>(FP, FT, acc);
    hist_kernel<<<24, 256, 0, stream>>>(pred, tgt, acc);

    // ---- VGG (pred) ----
    conv(VGGP, XA,  wpk_vgg[0], vgg_b[0],   3, 128, 128,  64, 1);
    conv(XA,   XB,  wpk_vgg[1], vgg_b[1],  64, 128, 128,  64, 1);
    pool(XB,   XA,  64, 128);
    conv(XA,   XB,  wpk_vgg[2], vgg_b[2],  64,  64,  64, 128, 1);
    conv(XB,   XA,  wpk_vgg[3], vgg_b[3], 128,  64,  64, 128, 1);
    pool(XA,   XB, 128, 64);
    conv(XB,   XA,  wpk_vgg[4], vgg_b[4], 128,  32,  32, 256, 1);
    conv(XA,   XB,  wpk_vgg[5], vgg_b[5], 256,  32,  32, 256, 1);
    conv(XB,   XA,  wpk_vgg[6], vgg_b[6], 256,  32,  32, 256, 1);
    conv(XA,   XB,  wpk_vgg[7], vgg_b[7], 256,  32,  32, 256, 1);
    pool(XB,   XA, 256, 32);
    conv(XA,   VFP, wpk_vgg[8], vgg_b[8], 256,  16,  16, 512, 1);
    // ---- VGG (target) ----
    conv(VGGT, XA,  wpk_vgg[0], vgg_b[0],   3, 128, 128,  64, 1);
    conv(XA,   XB,  wpk_vgg[1], vgg_b[1],  64, 128, 128,  64, 1);
    pool(XB,   XA,  64, 128);
    conv(XA,   XB,  wpk_vgg[2], vgg_b[2],  64,  64,  64, 128, 1);
    conv(XB,   XA,  wpk_vgg[3], vgg_b[3], 128,  64,  64, 128, 1);
    pool(XA,   XB, 128, 64);
    conv(XB,   XA,  wpk_vgg[4], vgg_b[4], 128,  32,  32, 256, 1);
    conv(XA,   XB,  wpk_vgg[5], vgg_b[5], 256,  32,  32, 256, 1);
    conv(XB,   XA,  wpk_vgg[6], vgg_b[6], 256,  32,  32, 256, 1);
    conv(XA,   XB,  wpk_vgg[7], vgg_b[7], 256,  32,  32, 256, 1);
    pool(XB,   XA, 256, 32);
    conv(XA,   XB,  wpk_vgg[8], vgg_b[8], 256,  16,  16, 512, 1);

    sqdiff_kernel<<<(524288 + 255) / 256, 256, 0, stream>>>(VFP, XB, acc, 524288);
    finalize_kernel<<<1, 1, 0, stream>>>(acc, (float*)d_out);
}
